// NeighborGatedAttention_6562710028717
// MI455X (gfx1250) — matrix operand order repaired and checked by reading, not compiled, not run on hardware
//
#include <hip/hip_runtime.h>

#define NNEI   128
#define EMBED  128
#define HIDDEN 128
#define HEADS  4
#define HEAD_DIM 32
#define ANGLE_DIM 64
#define N_ATOMS 2048
#define ATTNW_SHIFT 20.0f

typedef __attribute__((ext_vector_type(16))) __bf16 v16bf;
typedef __attribute__((ext_vector_type(8)))  __bf16 bf16x8;
typedef __attribute__((ext_vector_type(8)))  float  v8f;
typedef __attribute__((ext_vector_type(4)))  float  f32x4;
typedef __attribute__((ext_vector_type(4)))  unsigned int u32x4;
typedef __attribute__((ext_vector_type(8)))  int    i32x8;
typedef __attribute__((ext_vector_type(4)))  int    i32x4;

#ifndef __has_builtin
#define __has_builtin(x) 0
#endif
#if __has_builtin(__builtin_amdgcn_tensor_load_to_lds)
#define HAVE_TDM 1
#else
#define HAVE_TDM 0
#endif

// ---------------- LDS layout (bytes, dynamic) ----------------
#define SC_STRIDE 132                 // padded f32 row stride for scores/query staging
#define LDS_SC    0                   // 128*132*4 = 67584 : query f32 staging, then per-head scores
#define LDS_Q     67584               // 128*128 bf16
#define LDS_K     100352
#define LDS_V     133120
#define LDS_O     165888              // o (attn@v assembled) bf16 128*128
#define LDS_R     198656              // input_r padded [128][4] f32
#define LDS_S     200704              // s   [128] f32
#define LDS_SW    201216              // sw  [128] f32
#define LDS_BIN   201728              // b_in [384] f32
#define LDS_BOUT  203264              // b_out [128] f32
#define LDS_AW1   203776              // ae_w1 [6][64] f32
#define LDS_AB1   205312              // ae_b1 [64]
#define LDS_AW2   205568              // ae_w2 [64][4]
#define LDS_AB2   206592              // ae_b2 [4]
#define LDS_LNG   206608
#define LDS_LNB   206624
#define LDS_ASC   206640
#define LDS_MASK  206656              // nei_mask u8 [128]
#define LDS_TOTAL 206848

#define OUT_ATTN_OFF (2048u * 128u * 128u)   // output first, then attn, flat
#define WS_WOUT_OFF  49152                   // bf16 elements: w_in pack = 24*4*32*16
#define WS_NEED_BYTES ((49152 + 16384) * 2)

// ---- WMMA fragment loaders (contiguous k order, k = 16*(lane/16) + s), vectorized ----
// A: 16x32 16-bit. lanes 0-15: M=lane, K = 0..15; lanes 16-31: K = 16..31.
__device__ __forceinline__ v16bf ldA_f32(const float* src, int stride, int row0, int k0, int lane) {
  const float* p = src + (row0 + (lane & 15)) * stride + k0 + ((lane & 16) ? 16 : 0);
  f32x4 x0 = *(const f32x4*)(p + 0);
  f32x4 x1 = *(const f32x4*)(p + 4);
  f32x4 x2 = *(const f32x4*)(p + 8);
  f32x4 x3 = *(const f32x4*)(p + 12);
  v16bf a;
#pragma unroll
  for (int j = 0; j < 4; ++j) {
    a[j]      = (__bf16)x0[j];
    a[4 + j]  = (__bf16)x1[j];
    a[8 + j]  = (__bf16)x2[j];
    a[12 + j] = (__bf16)x3[j];
  }
  return a;
}
__device__ __forceinline__ v16bf ldA_bf16(const __bf16* src, int stride, int row0, int k0, int lane) {
  const __bf16* p = src + (row0 + (lane & 15)) * stride + k0 + ((lane & 16) ? 16 : 0);
  bf16x8 lo = *(const bf16x8*)(p);
  bf16x8 hi = *(const bf16x8*)(p + 8);
  return __builtin_shufflevector(lo, hi, 0, 1, 2, 3, 4, 5, 6, 7, 8, 9, 10, 11, 12, 13, 14, 15);
}
// B: 32x16 16-bit, source stored [K][N] row-major. lane = column; lanes 0-15 K=0..15, 16-31 K=16..31.
__device__ __forceinline__ v16bf ldB_f32(const float* src, int stride, int k0, int n0, int lane) {
  const float* p = src + (size_t)(k0 + ((lane & 16) ? 16 : 0)) * stride + n0 + (lane & 15);
  v16bf b;
#pragma unroll
  for (int j = 0; j < 16; ++j) b[j] = (__bf16)p[(size_t)j * stride];
  return b;
}
__device__ __forceinline__ v16bf ldB_bf16(const __bf16* src, int stride, int k0, int n0, int lane) {
  const __bf16* p = src + (size_t)(k0 + ((lane & 16) ? 16 : 0)) * stride + n0 + (lane & 15);
  v16bf b;
#pragma unroll
  for (int j = 0; j < 16; ++j) b[j] = p[(size_t)j * stride];
  return b;
}
// B where the source is stored transposed [N][K] row-major (contiguous 32B per lane).
__device__ __forceinline__ v16bf ldBt_bf16(const __bf16* src, int stride, int k0, int n0, int lane) {
  const __bf16* p = src + (size_t)(n0 + (lane & 15)) * stride + k0 + ((lane & 16) ? 16 : 0);
  return *(const v16bf*)p;
}
// Pre-packed B fragment: one coalesced 32B load per lane.
__device__ __forceinline__ v16bf ldB_pk(const __bf16* ws, int tile, int lane) {
  return *(const v16bf*)(ws + (((size_t)tile * 32 + lane) << 4));
}

#define WMMA_BF16(c, a, b) \
  (c) = __builtin_amdgcn_wmma_f32_16x16x32_bf16(false, (a), false, (b), (short)0, (c), false, false)

// -------- prep: pack w_in / w_out into per-lane WMMA-B bf16 fragments in d_ws --------
__global__ void __launch_bounds__(256)
pack_weights_kernel(const float* __restrict__ w_in, const float* __restrict__ w_out,
                    __bf16* __restrict__ ws) {
  const int t = blockIdx.x * 256 + threadIdx.x;
  if (t < 24 * 4 * 32) {                      // w_in: 24 n-tiles x 4 k-steps
    const int lane = t & 31, tile = t >> 5;
    const int kk = tile & 3, nt = tile >> 2;
    const int n  = nt * 16 + (lane & 15);
    const int kb = kk * 32 + ((lane & 16) ? 16 : 0);
    __bf16* dst = ws + ((size_t)t << 4);
#pragma unroll
    for (int j = 0; j < 16; ++j) dst[j] = (__bf16)w_in[(size_t)(kb + j) * 384 + n];
  } else if (t < 4096) {                      // w_out: 8 n-tiles x 4 k-steps
    const int t2 = t - 3072;
    const int lane = t2 & 31, tile = t2 >> 5;
    const int kk = tile & 3, nt = tile >> 2;
    const int n  = nt * 16 + (lane & 15);
    const int kb = kk * 32 + ((lane & 16) ? 16 : 0);
    __bf16* dst = ws + WS_WOUT_OFF + ((size_t)t2 << 4);
#pragma unroll
    for (int j = 0; j < 16; ++j) dst[j] = (__bf16)w_out[(size_t)(kb + j) * 128 + n];
  }
}

__global__ void __launch_bounds__(256)
nga_fused_kernel(const float* __restrict__ query, const unsigned char* __restrict__ nei_mask,
                 const float* __restrict__ input_r, const float* __restrict__ sw,
                 const float* __restrict__ s, const int* __restrict__ atype,
                 const float* __restrict__ w_in, const float* __restrict__ b_in,
                 const float* __restrict__ w_out, const float* __restrict__ b_out,
                 const float* __restrict__ ae_w1, const float* __restrict__ ae_b1,
                 const float* __restrict__ ae_w2, const float* __restrict__ ae_b2,
                 const float* __restrict__ ln_g, const float* __restrict__ ln_b,
                 const float* __restrict__ angle_scale, float* __restrict__ out,
                 const __bf16* __restrict__ wpk, int use_pk) {
  extern __shared__ char sm[];
  const int t    = threadIdx.x;
  const int lane = t & 31;
  const int wave = t >> 5;
  const int atom = blockIdx.x;

  float* qf   = (float*)(sm + LDS_SC);      // query f32 staging / scores
  __bf16* qL  = (__bf16*)(sm + LDS_Q);
  __bf16* kL  = (__bf16*)(sm + LDS_K);
  __bf16* vL  = (__bf16*)(sm + LDS_V);
  __bf16* oL  = (__bf16*)(sm + LDS_O);
  float* rbuf = (float*)(sm + LDS_R);
  float* sbuf = (float*)(sm + LDS_S);
  float* swb  = (float*)(sm + LDS_SW);
  float* binb = (float*)(sm + LDS_BIN);
  float* boutb= (float*)(sm + LDS_BOUT);
  float* aw1  = (float*)(sm + LDS_AW1);
  float* ab1  = (float*)(sm + LDS_AB1);
  float* aw2  = (float*)(sm + LDS_AW2);
  float* ab2  = (float*)(sm + LDS_AB2);
  float* lng  = (float*)(sm + LDS_LNG);
  float* lnb  = (float*)(sm + LDS_LNB);
  float* ascl = (float*)(sm + LDS_ASC);
  unsigned char* mskb = (unsigned char*)(sm + LDS_MASK);

  __builtin_prefetch(w_in, 0, 1);
  __builtin_prefetch(w_out, 0, 1);

  // -------- phase 0: small per-atom / weight vectors to LDS --------
  if (t < 128) {
    rbuf[t * 4 + 0] = input_r[(size_t)atom * 384 + t * 3 + 0];
    rbuf[t * 4 + 1] = input_r[(size_t)atom * 384 + t * 3 + 1];
    rbuf[t * 4 + 2] = input_r[(size_t)atom * 384 + t * 3 + 2];
    rbuf[t * 4 + 3] = 0.0f;
    sbuf[t] = s[(size_t)atom * 128 + t];
    swb[t]  = sw[(size_t)atom * 128 + t];
    mskb[t] = nei_mask[(size_t)atom * 128 + t];
    boutb[t] = b_out[t];
  }
  for (int i = t; i < 384; i += 256) { binb[i] = b_in[i]; aw1[i] = ae_w1[i]; }
  if (t < 256) aw2[t] = ae_w2[t];
  if (t < 64) ab1[t] = ae_b1[t];
  if (t < 4) { ab2[t] = ae_b2[t]; lng[t] = ln_g[t]; lnb[t] = ln_b[t]; }
  if (t == 0) ascl[0] = angle_scale[0];
  const int at = atype[atom];
  const int kval = (at == 0) ? 32 : (at == 1) ? 48 : (at == 2) ? 64 : 128;

  // -------- phase 1: stage query f32 into padded LDS via TDM (tensor DMA) --------
  {
    const float* qsrc = query + (size_t)atom * (128 * 128);
#if HAVE_TDM
    if (wave == 0) {
      // D# group 0: count=1, lds_addr, global_addr[56:0], type=2 ("image")
      const unsigned long long ga = (unsigned long long)(uintptr_t)qsrc;
      u32x4 g0;
      g0[0] = 1u;
      g0[1] = (unsigned int)(uintptr_t)qf;
      g0[2] = (unsigned int)(ga & 0xffffffffu);
      g0[3] = (unsigned int)((ga >> 32) & 0x01ffffffu) | 0x80000000u;
      // D# group 1: data_size=4B, pad_enable, pad every 128 DW by 4 DW (-> stride 132),
      // tensor_dim0=128, tensor_dim1=128, tile 128x128, dim0_stride=128
      i32x8 g1;
      g1[0] = 0x07920000;            // pad_amount=3(4DW) | pad_interval=6(128DW) | pad_en | data_size=2
      g1[1] = (int)(128u << 16);     // tensor_dim0[15:0] in bits[31:16]
      g1[2] = (int)(128u << 16);     // tensor_dim0 hi=0, tensor_dim1[15:0] in bits[31:16]
      g1[3] = (int)(128u << 16);     // tensor_dim1 hi=0, tile_dim0=128 in bits[31:16]
      g1[4] = 128;                   // tile_dim1=128, tile_dim2=0
      g1[5] = 128;                   // tensor_dim0_stride low32
      g1[6] = 0;                     // stride hi, tensor_dim1_stride (unused 2D)
      g1[7] = 0;
      i32x4 z4 = {0, 0, 0, 0};
#if __clang_major__ >= 23
      i32x8 z8 = {0, 0, 0, 0, 0, 0, 0, 0};
      __builtin_amdgcn_tensor_load_to_lds(g0, g1, z4, z4, z8, 0);
#else
      __builtin_amdgcn_tensor_load_to_lds(g0, g1, z4, z4, 0);
#endif
      __builtin_amdgcn_s_wait_tensorcnt(0);
    }
#else
    for (int idx = t; idx < 128 * 32; idx += 256) {
      int i = idx >> 5, c4 = (idx & 31) << 2;
      f32x4 v = *(const f32x4*)(qsrc + i * 128 + c4);
      *(f32x4*)(qf + i * SC_STRIDE + c4) = v;
    }
#endif
  }
  __syncthreads();

  // -------- GEMM1: qkv = query @ w_in + b_in  (bf16 WMMA, f32 acc) --------
  for (int nn = 0; nn < 3; ++nn) {
    const int nt = wave * 3 + nn;          // 24 n-tiles over 8 waves
    const int cbase = nt * 16;
    v16bf B0, B1, B2, B3;
    if (use_pk) {
      B0 = ldB_pk(wpk, nt * 4 + 0, lane);
      B1 = ldB_pk(wpk, nt * 4 + 1, lane);
      B2 = ldB_pk(wpk, nt * 4 + 2, lane);
      B3 = ldB_pk(wpk, nt * 4 + 3, lane);
    } else {
      B0 = ldB_f32(w_in, 384, 0,  cbase, lane);
      B1 = ldB_f32(w_in, 384, 32, cbase, lane);
      B2 = ldB_f32(w_in, 384, 64, cbase, lane);
      B3 = ldB_f32(w_in, 384, 96, cbase, lane);
    }
    const int cg  = cbase + (lane & 15);
    const float bc = binb[cg];
    __bf16* dst = (__bf16*)(sm + LDS_Q + (cg >> 7) * 32768);
    const int col = cg & 127;
    for (int mt = 0; mt < 8; ++mt) {
      v8f C = {};
      WMMA_BF16(C, ldA_f32(qf, SC_STRIDE, mt * 16, 0,  lane), B0);
      WMMA_BF16(C, ldA_f32(qf, SC_STRIDE, mt * 16, 32, lane), B1);
      WMMA_BF16(C, ldA_f32(qf, SC_STRIDE, mt * 16, 64, lane), B2);
      WMMA_BF16(C, ldA_f32(qf, SC_STRIDE, mt * 16, 96, lane), B3);
#pragma unroll
      for (int r = 0; r < 8; ++r) {
        int row = mt * 16 + r + ((lane & 16) ? 8 : 0);
        dst[row * 128 + col] = (__bf16)(C[r] + bc);
      }
    }
  }
  __syncthreads();

  // -------- phase 1.5: per-(row,head) L2 norm of q,k,v; q *= HEAD_DIM^-0.5 --------
  for (int idx = t; idx < 128 * 4 * 3; idx += 256) {
    int mat = idx / 512, rem = idx - mat * 512;
    int i = rem >> 2, h = rem & 3;
    __bf16* p = (__bf16*)(sm + LDS_Q + mat * 32768) + i * 128 + h * 32;
    float ss = 0.0f;
#pragma unroll
    for (int d = 0; d < 32; ++d) { float x = (float)p[d]; ss = fmaf(x, x, ss); }
    float inv = rsqrtf(fmaxf(ss, 1e-24f));
    if (mat == 0) inv *= 0.17677669529663689f;   // 1/sqrt(32)
#pragma unroll
    for (int d = 0; d < 32; ++d) p[d] = (__bf16)((float)p[d] * inv);
  }

  // -------- phase 2: angle-MLP bias (symmetric i<=j), stash into attn out region --------
  float* attn_out = out + OUT_ATTN_OFF + (size_t)atom * (4 * 128 * 128);
  const float ascv = ascl[0];
  for (int p = t; p < 8256; p += 256) {
    int i = (int)(128.5f - sqrtf(fmaxf(128.5f * 128.5f - 2.0f * (float)p, 0.0f)));
    if (i < 0) i = 0; if (i > 127) i = 127;
    while (i > 0 && (i * 128 - (i * (i - 1)) / 2) > p) --i;
    while (i < 127 && ((i + 1) * 128 - ((i + 1) * i) / 2) <= p) ++i;
    const int j = i + (p - (i * 128 - (i * (i - 1)) / 2));
    const float* ri = rbuf + i * 4; const float* rj = rbuf + j * 4;
    float c = ri[0] * rj[0] + ri[1] * rj[1] + ri[2] * rj[2];
    c = fminf(1.0f, fmaxf(-1.0f, c));
    const float sn = sqrtf(1.0f - c * c + 1e-8f);
    const float f0 = c, f1 = sn, f2 = 2.0f * sn * c, f3 = __expf(2.0f * (c - 1.0f));
    const float si = sbuf[i], sj = sbuf[j];
    const float f4 = si + sj, f5 = fabsf(si - sj);
    float a0 = ab2[0], a1 = ab2[1], a2 = ab2[2], a3 = ab2[3];
    for (int cc = 0; cc < 64; ++cc) {
      float hh = ab1[cc];
      hh = fmaf(f0, aw1[cc],        hh);
      hh = fmaf(f1, aw1[64 + cc],   hh);
      hh = fmaf(f2, aw1[128 + cc],  hh);
      hh = fmaf(f3, aw1[192 + cc],  hh);
      hh = fmaf(f4, aw1[256 + cc],  hh);
      hh = fmaf(f5, aw1[320 + cc],  hh);
      const float act = hh / (1.0f + __expf(-hh));      // silu
      a0 = fmaf(act, aw2[cc * 4 + 0], a0);
      a1 = fmaf(act, aw2[cc * 4 + 1], a1);
      a2 = fmaf(act, aw2[cc * 4 + 2], a2);
      a3 = fmaf(act, aw2[cc * 4 + 3], a3);
    }
    const float mu = 0.25f * (a0 + a1 + a2 + a3);
    const float d0 = a0 - mu, d1 = a1 - mu, d2 = a2 - mu, d3 = a3 - mu;
    const float rstd = rsqrtf(0.25f * (d0 * d0 + d1 * d1 + d2 * d2 + d3 * d3) + 1e-5f);
    const bool inm = (i < kval) && (j < kval);
    float bh[4] = { d0, d1, d2, d3 };
#pragma unroll
    for (int h = 0; h < 4; ++h) {
      float b = (bh[h] * rstd * lng[h] + lnb[h]) * ascv;
      if (!inm) b = 0.0f;
      attn_out[(h * 128 + i) * 128 + j] = b;
      attn_out[(h * 128 + j) * 128 + i] = b;
    }
  }
  __threadfence();
  __syncthreads();

  // -------- phase 3/4: per head: scores -> gate -> softmax -> attn out; o = attn @ v --------
  for (int h = 0; h < 4; ++h) {
    // 3a: q_h @ k_h^T; wave == m-tile
    {
      const int mt = wave;
      v16bf Aq = ldA_bf16(qL, 128, mt * 16, h * 32, lane);
      for (int nt = 0; nt < 8; ++nt) {
        v16bf Bk = ldBt_bf16(kL, 128, h * 32, nt * 16, lane);
        v8f C = {};
        WMMA_BF16(C, Aq, Bk);
        const int j = nt * 16 + (lane & 15);
        const float swk = swb[j];
        const float* bptr = attn_out + (h * 128) * 128 + j;
#pragma unroll
        for (int r = 0; r < 8; ++r) {
          int i = mt * 16 + r + ((lane & 16) ? 8 : 0);
          float val = (C[r] + bptr[i * 128] + ATTNW_SHIFT) * swb[i] * swk - ATTNW_SHIFT;
          qf[i * SC_STRIDE + j] = val;
        }
      }
    }
    __syncthreads();
    // 3c: row softmax + gating, write final attn to global, keep in LDS for phase 4
    if (t < 128) {
      const int i = t;
      float* row = qf + i * SC_STRIDE;
      float mx = -1e30f;
      for (int j = 0; j < 128; ++j) mx = fmaxf(mx, row[j]);
      float sum = 0.0f;
      for (int j = 0; j < 128; ++j) { float e = __expf(row[j] - mx); row[j] = e; sum += e; }
      const float fi = swb[i] * (1.0f / sum) * (mskb[i] ? 1.0f : 0.0f);
      float* aout = attn_out + (h * 128 + i) * 128;
      for (int j = 0; j < 128; ++j) {
        float a = row[j] * fi * swb[j];
        row[j] = a;
        aout[j] = a;
      }
    }
    __syncthreads();
    // 4: o_h = attn @ v_h ; wave == m-tile, 2 n-tiles of 16 over HEAD_DIM
    {
      const int mt = wave;
      v16bf A0 = ldA_f32(qf, SC_STRIDE, mt * 16, 0,  lane);
      v16bf A1 = ldA_f32(qf, SC_STRIDE, mt * 16, 32, lane);
      v16bf A2 = ldA_f32(qf, SC_STRIDE, mt * 16, 64, lane);
      v16bf A3 = ldA_f32(qf, SC_STRIDE, mt * 16, 96, lane);
      for (int nt = 0; nt < 2; ++nt) {
        const int dbase = h * 32 + nt * 16;
        v8f C = {};
        WMMA_BF16(C, A0, ldB_bf16(vL, 128, 0,  dbase, lane));
        WMMA_BF16(C, A1, ldB_bf16(vL, 128, 32, dbase, lane));
        WMMA_BF16(C, A2, ldB_bf16(vL, 128, 64, dbase, lane));
        WMMA_BF16(C, A3, ldB_bf16(vL, 128, 96, dbase, lane));
        const int col = dbase + (lane & 15);
#pragma unroll
        for (int r = 0; r < 8; ++r) {
          int i = mt * 16 + r + ((lane & 16) ? 8 : 0);
          oL[i * 128 + col] = (__bf16)C[r];
        }
      }
    }
    __syncthreads();
  }

  // -------- phase 5: output = o @ w_out + b_out --------
  {
    float* outp = out + (size_t)atom * (128 * 128);
    const int mt = wave;
    v16bf A0 = ldA_bf16(oL, 128, mt * 16, 0,  lane);
    v16bf A1 = ldA_bf16(oL, 128, mt * 16, 32, lane);
    v16bf A2 = ldA_bf16(oL, 128, mt * 16, 64, lane);
    v16bf A3 = ldA_bf16(oL, 128, mt * 16, 96, lane);
    for (int nt = 0; nt < 8; ++nt) {
      const int cb = nt * 16;
      v8f C = {};
      if (use_pk) {
        const __bf16* wp = wpk + WS_WOUT_OFF;
        WMMA_BF16(C, A0, ldB_pk(wp, nt * 4 + 0, lane));
        WMMA_BF16(C, A1, ldB_pk(wp, nt * 4 + 1, lane));
        WMMA_BF16(C, A2, ldB_pk(wp, nt * 4 + 2, lane));
        WMMA_BF16(C, A3, ldB_pk(wp, nt * 4 + 3, lane));
      } else {
        WMMA_BF16(C, A0, ldB_f32(w_out, 128, 0,  cb, lane));
        WMMA_BF16(C, A1, ldB_f32(w_out, 128, 32, cb, lane));
        WMMA_BF16(C, A2, ldB_f32(w_out, 128, 64, cb, lane));
        WMMA_BF16(C, A3, ldB_f32(w_out, 128, 96, cb, lane));
      }
      const int cc = cb + (lane & 15);
      const float bo = boutb[cc];
#pragma unroll
      for (int r = 0; r < 8; ++r) {
        int i = mt * 16 + r + ((lane & 16) ? 8 : 0);
        outp[i * 128 + cc] = C[r] + bo;
      }
    }
  }
}

extern "C" void kernel_launch(void* const* d_in, const int* in_sizes, int n_in,
                              void* d_out, int out_size, void* d_ws, size_t ws_size,
                              hipStream_t stream) {
  (void)in_sizes; (void)n_in; (void)out_size;
  const float*         query      = (const float*)d_in[0];
  const unsigned char* nmask      = (const unsigned char*)d_in[1];
  const float*         input_r    = (const float*)d_in[2];
  const float*         sw         = (const float*)d_in[3];
  const float*         s          = (const float*)d_in[4];
  const int*           atype      = (const int*)d_in[5];
  const float*         w_in       = (const float*)d_in[6];
  const float*         b_in       = (const float*)d_in[7];
  const float*         w_out      = (const float*)d_in[8];
  const float*         b_out      = (const float*)d_in[9];
  const float*         ae_w1      = (const float*)d_in[10];
  const float*         ae_b1      = (const float*)d_in[11];
  const float*         ae_w2      = (const float*)d_in[12];
  const float*         ae_b2      = (const float*)d_in[13];
  const float*         ln_g       = (const float*)d_in[14];
  const float*         ln_b       = (const float*)d_in[15];
  const float*         angle_s    = (const float*)d_in[16];
  float* out = (float*)d_out;

  const int use_pk = (d_ws != nullptr && ws_size >= (size_t)WS_NEED_BYTES) ? 1 : 0;
  __bf16* wpk = (__bf16*)d_ws;
  if (use_pk) {
    pack_weights_kernel<<<16, 256, 0, stream>>>(w_in, w_out, wpk);
  }
  nga_fused_kernel<<<N_ATOMS, 256, LDS_TOTAL, stream>>>(
      query, nmask, input_r, sw, s, atype, w_in, b_in, w_out, b_out,
      ae_w1, ae_b1, ae_w2, ae_b2, ln_g, ln_b, angle_s, out, wpk, use_pk);
}
